// GroupQuantLinear_18382460026880
// MI455X (gfx1250) — matrix operand order repaired and checked by reading, not compiled, not run on hardware
//
#include <hip/hip_runtime.h>
#include <stdint.h>

#define TOKENS 8192
#define INF    4096
#define OUTF   11008
#define BM 128
#define BN 128
#define BK 64
#define LDSP 72                 // halfs per LDS tile row = 144B (TDM-padded)
#define KSTEPS (INF / BK)       // 64

typedef _Float16 v16h __attribute__((ext_vector_type(16)));
typedef _Float16 v8h  __attribute__((ext_vector_type(8)));
typedef _Float16 v4h  __attribute__((ext_vector_type(4)));
typedef float    v8f  __attribute__((ext_vector_type(8)));
typedef unsigned int u32x4 __attribute__((ext_vector_type(4)));
typedef int      i32x8 __attribute__((ext_vector_type(8)));
typedef int      i32x4 __attribute__((ext_vector_type(4)));

// ---------------- prepass 1: activations f32 -> f16 ----------------
__global__ __launch_bounds__(256) void cvt_x_kernel(const float4* __restrict__ x,
                                                    v4h* __restrict__ xh) {
  unsigned tid = blockIdx.x * 256u + threadIdx.x;
  float4 v = x[tid];
  v4h h;
  h[0] = (_Float16)v.x; h[1] = (_Float16)v.y;
  h[2] = (_Float16)v.z; h[3] = (_Float16)v.w;
  xh[tid] = h;
}

// ---------------- prepass 2: int4 group dequant -> f16 weights ----------------
// w_packed (out, 64 groups, 16 ids) int32 holding uint16; big-endian nibbles.
// Output offset works out to exactly tid*4 halfs (contiguous).
__global__ __launch_bounds__(256) void dequant_w_kernel(const int* __restrict__ wp,
                                                        const float* __restrict__ ws,
                                                        const float* __restrict__ wb,
                                                        v4h* __restrict__ wh) {
  unsigned tid = blockIdx.x * 256u + threadIdx.x;
  int p = wp[tid];
  float s = ws[tid >> 4];
  float b = wb[tid >> 4];
  v4h h;
  h[0] = (_Float16)((float)((p >> 12) & 15) * s + b);
  h[1] = (_Float16)((float)((p >> 8)  & 15) * s + b);
  h[2] = (_Float16)((float)((p >> 4)  & 15) * s + b);
  h[3] = (_Float16)((float)( p        & 15) * s + b);
  wh[tid] = h;
}

// ---------------- TDM: DMA one 128x64 f16 tile (row stride INF) into LDS ----------------
// D# group1: data_size=2B, pad_enable, pad_interval=32 dwords (128B row), pad_amount=4 dwords
// -> LDS row pitch 144B, bank-conflict-free b128 fragment loads.
__device__ __forceinline__ void tdm_load_tile(uint32_t lds_off, const _Float16* gp,
                                              uint32_t nrows) {
  uint64_t ga = (uint64_t)(uintptr_t)gp;
  u32x4 g0;
  g0[0] = 1u;                                                  // count=1, user descriptor
  g0[1] = lds_off;                                             // LDS byte address
  g0[2] = (uint32_t)ga;                                        // global addr lo
  g0[3] = (uint32_t)((ga >> 32) & 0x1FFFFFFull) | (2u << 30);  // addr hi | type=2 (image)
  i32x8 g1;
  g1[0] = (int)((1u << 16) | (1u << 20) | (4u << 22) | (3u << 25)); // dsize=2B|pad|int32dw|amt4dw
  g1[1] = (int)(((unsigned)INF & 0xFFFFu) << 16);              // tensor_dim0 lo
  g1[2] = (int)((((unsigned)INF >> 16) & 0xFFFFu) | ((nrows & 0xFFFFu) << 16)); // dim0 hi|dim1 lo
  g1[3] = (int)(((nrows >> 16) & 0xFFFFu) | ((unsigned)BK << 16));  // dim1 hi | tile_dim0=64
  g1[4] = (int)BM;                                             // tile_dim1=128, tile_dim2=0
  g1[5] = (int)INF;                                            // tensor_dim0_stride lo
  g1[6] = 0;                                                   // stride0 hi | stride1 lo
  g1[7] = 0;
  i32x4 gz4 = {0, 0, 0, 0};
  i32x8 gz8 = {0, 0, 0, 0, 0, 0, 0, 0};
  __builtin_amdgcn_tensor_load_to_lds(g0, g1, gz4, gz4, gz8, 0);
}

// ---------------- main GEMM: C = Xh * Wh^T + b ----------------
__global__ __launch_bounds__(256) void gemm_kernel(const _Float16* __restrict__ xh,
                                                   const _Float16* __restrict__ wh,
                                                   const float* __restrict__ bias,
                                                   float* __restrict__ out) {
  __shared__ __align__(16) _Float16 lds[2][2][BM * LDSP];      // 73,728 B

  const unsigned tid  = threadIdx.x;
  const unsigned wave = tid >> 5, lane = tid & 31;
  const unsigned wm = wave >> 2, wn = wave & 3;                // 2 x 4 wave grid
  const unsigned l16 = lane & 15, lh = lane >> 4;

  const unsigned blockN = blockIdx.x * BN;
  const unsigned blockM = blockIdx.y * BM;

  const _Float16* xtile = xh + (size_t)blockM * INF;
  const _Float16* wtile = wh + (size_t)blockN * INF;

  v8f acc[4][2] = {};

  if (wave == 0) {
    tdm_load_tile((uint32_t)(uintptr_t)&lds[0][0][0], xtile, TOKENS);
    tdm_load_tile((uint32_t)(uintptr_t)&lds[0][1][0], wtile, OUTF);
  }

#pragma unroll 1
  for (int t = 0; t < KSTEPS; ++t) {
    const int cur = t & 1, nxt = cur ^ 1;
    __syncthreads();                       // everyone done reading buf[nxt] (iter t-1)
    if (wave == 0) {
      if (t + 1 < KSTEPS) {
        tdm_load_tile((uint32_t)(uintptr_t)&lds[nxt][0][0], xtile + (t + 1) * BK, TOKENS);
        tdm_load_tile((uint32_t)(uintptr_t)&lds[nxt][1][0], wtile + (t + 1) * BK, OUTF);
        __builtin_amdgcn_s_wait_tensorcnt(2);   // tile t landed (in-order per wave)
      } else {
        __builtin_amdgcn_s_wait_tensorcnt(0);
      }
    }
    __syncthreads();                       // tile t visible to all waves

    const _Float16* xb = &lds[cur][0][0];
    const _Float16* wb = &lds[cur][1][0];
#pragma unroll
    for (int kk = 0; kk < BK; kk += 32) {
      v16h a[4], bf[2];
#pragma unroll
      for (int fm = 0; fm < 4; ++fm) {
        // A 16x32 f16: lanes 0-15 row=l16 K={0..7,16..23}; lanes 16-31 K={8..15,24..31}
        const v8h* p = (const v8h*)(xb + (wm * 64 + fm * 16 + l16) * LDSP + kk + lh * 8);
        a[fm] = __builtin_shufflevector(p[0], p[2],
                    0,1,2,3,4,5,6,7,8,9,10,11,12,13,14,15);
      }
#pragma unroll
      for (int fn = 0; fn < 2; ++fn) {
        // B 32x16 f16: lane n=l16; lanes 0-15 K={0..7,16..23}, lanes 16-31 K={8..15,24..31}
        const v8h* q = (const v8h*)(wb + (wn * 32 + fn * 16 + l16) * LDSP + kk + lh * 8);
        bf[fn] = __builtin_shufflevector(q[0], q[2],
                    0,1,2,3,4,5,6,7,8,9,10,11,12,13,14,15);
      }
#pragma unroll
      for (int fm = 0; fm < 4; ++fm)
#pragma unroll
        for (int fn = 0; fn < 2; ++fn)
          acc[fm][fn] = __builtin_amdgcn_wmma_f32_16x16x32_f16(
              false, a[fm], false, bf[fn], (short)0, acc[fm][fn], false, false);
    }
  }

  // epilogue: C/D layout — VGPR r, lanes 0-15: M=r, N=l16; lanes 16-31: M=r+8
#pragma unroll
  for (int fn = 0; fn < 2; ++fn) {
    const unsigned col = blockN + wn * 32 + fn * 16 + l16;
    const float bv = bias[col];
#pragma unroll
    for (int fm = 0; fm < 4; ++fm) {
      const unsigned row0 = blockM + wm * 64 + fm * 16 + lh * 8;
#pragma unroll
      for (int r = 0; r < 8; ++r)
        out[(size_t)(row0 + r) * OUTF + col] = acc[fm][fn][r] + bv;
    }
  }
}

extern "C" void kernel_launch(void* const* d_in, const int* in_sizes, int n_in,
                              void* d_out, int out_size, void* d_ws, size_t ws_size,
                              hipStream_t stream) {
  const float* x      = (const float*)d_in[0];
  const int*   wp     = (const int*)d_in[1];
  const float* wscale = (const float*)d_in[2];
  const float* wbias  = (const float*)d_in[3];
  const float* b      = (const float*)d_in[4];
  float* out = (float*)d_out;

  _Float16* xh = (_Float16*)d_ws;                                   // 67,108,864 B
  _Float16* wh = (_Float16*)((char*)d_ws +
                             (size_t)TOKENS * INF * sizeof(_Float16)); // +90,177,536 B

  cvt_x_kernel<<<(TOKENS * (size_t)INF) / 4 / 256, 256, 0, stream>>>((const float4*)x, (v4h*)xh);
  dequant_w_kernel<<<((size_t)OUTF * 64 * 16) / 256, 256, 0, stream>>>(wp, wscale, wbias, (v4h*)wh);
  gemm_kernel<<<dim3(OUTF / BN, TOKENS / BM), 256, 0, stream>>>(xh, wh, b, out);
}
